// MemoryCrossAttention_33243046871174
// MI455X (gfx1250) — matrix operand order repaired and checked by reading, not compiled, not run on hardware
//
#include <hip/hip_runtime.h>
#include <hip/hip_bf16.h>
#include <stdint.h>

typedef __attribute__((ext_vector_type(16))) __bf16 v16bf;
typedef __attribute__((ext_vector_type(8)))  float  v8f;
typedef unsigned short u16;
typedef unsigned char  u8;

static constexpr int Hc  = 2048;   // hidden
static constexpr int NHc = 16;     // heads
static constexpr int HDc = 128;    // head dim
static constexpr int Mc  = 256;    // memory tokens
static constexpr int Lc  = 4096;   // seq len
static constexpr int Bc  = 4;      // batch
static constexpr float ATT_SCALE = 0.08838834764831845f; // 128^-0.5

__device__ __forceinline__ u16 f2bf(float f) {
  union { float f; uint32_t u; } v; v.f = f;
  uint32_t u = v.u + 0x7FFFu + ((v.u >> 16) & 1u);  // round-to-nearest-even
  return (u16)(u >> 16);
}
__device__ __forceinline__ float bf2f(u16 h) {
  union { uint32_t u; float f; } v; v.u = ((uint32_t)h) << 16;
  return v.f;
}

union Frag { v16bf v; uint4 q[2]; };

__device__ __forceinline__ v8f vzero() {
  v8f z;
  #pragma unroll
  for (int i = 0; i < 8; ++i) z[i] = 0.f;
  return z;
}

// ---- CDNA5 async global->LDS (ASYNCcnt-tracked, §15.18.3 op 98) -------------
// vdst = 32-bit LDS byte address (low 32 bits of shared-aperture address),
// vaddr = 64-bit global address. 16 B per lane per instruction.
__device__ __forceinline__ void async_g2l_b128(uint32_t lds_addr, const void* gaddr) {
  asm volatile("global_load_async_to_lds_b128 %0, %1, off"
               :
               : "v"(lds_addr), "v"(gaddr)
               : "memory");
}
__device__ __forceinline__ void wait_async_le4() {
  asm volatile("s_wait_asynccnt 0x4" ::: "memory");
}
__device__ __forceinline__ void wait_async_le0() {
  asm volatile("s_wait_asynccnt 0x0" ::: "memory");
}
__device__ __forceinline__ uint32_t lds_addr_of(const void* p) {
  return (uint32_t)(uintptr_t)p;   // LDS aperture: offset lives in addr[31:0]
}

// A-matrix fragment (16x32 bf16, row-major source, K contiguous).
// ISA 7.12.2: lanes 0-15 hold K {kb..kb+7, kb+16..kb+23} with kb=0, lanes 16-31 kb=8.
__device__ __forceinline__ Frag load_a_frag(const u16* base, int ld, int lane) {
  const int row = lane & 15;
  const int kb  = (lane < 16) ? 0 : 8;
  Frag f;
  f.q[0] = *(const uint4*)(base + (size_t)row * ld + kb);
  f.q[1] = *(const uint4*)(base + (size_t)row * ld + kb + 16);
  return f;
}
// B-matrix fragment (32x16 bf16). Source is N-major with K contiguous per row
// (i.e. src[n*ld + k] == B[k][n]). Lanes 0-15: K {0..7,16..23} of col=lane; 16-31: K {8..15,24..31}.
__device__ __forceinline__ Frag load_b_frag(const u16* base, int ld, int lane) {
  const int col = lane & 15;
  const int kb  = (lane < 16) ? 0 : 8;
  Frag f;
  f.q[0] = *(const uint4*)(base + (size_t)col * ld + kb);
  f.q[1] = *(const uint4*)(base + (size_t)col * ld + kb + 16);
  return f;
}

__device__ __forceinline__ v8f wmma_bf16(const Frag& a, const Frag& b, v8f c) {
  return __builtin_amdgcn_wmma_f32_16x16x32_bf16(false, a.v, false, b.v,
                                                 (short)0, c, false, false);
}

// ---------------------------------------------------------------- converters
__global__ __launch_bounds__(256) void cvt_bf16_kernel(const float* __restrict__ in,
                                                       u16* __restrict__ out, int n) {
  for (int i = blockIdx.x * 256 + threadIdx.x; i < n; i += gridDim.x * 256)
    out[i] = f2bf(in[i]);
}

// ---------------------------------------------------------------- rmsnorm
__global__ __launch_bounds__(256) void rmsnorm_kernel(const float* __restrict__ X,
                                                      const float* __restrict__ Wn,
                                                      u16* __restrict__ Out) {
  const int row = blockIdx.x;
  const int tid = threadIdx.x;
  const float* xr = X + (size_t)row * Hc;
  float ss = 0.f;
  for (int i = tid; i < Hc; i += 256) { float v = xr[i]; ss += v * v; }
  #pragma unroll
  for (int o = 16; o > 0; o >>= 1) ss += __shfl_xor(ss, o);
  __shared__ float red[8];
  if ((tid & 31) == 0) red[tid >> 5] = ss;
  __syncthreads();
  float tot = 0.f;
  #pragma unroll
  for (int i = 0; i < 8; ++i) tot += red[i];
  const float sc = rsqrtf(tot * (1.0f / (float)Hc) + 1e-6f);
  for (int i = tid; i < Hc; i += 256)
    Out[(size_t)row * Hc + i] = f2bf(xr[i] * sc * Wn[i]);
}

// ---------------------------------------------------------------- GEMM
// C[M,N] = A[M,K] @ W[N,K]^T  (nn.Linear).  Block tile 128x128, 8 waves,
// each wave 32x64 (2x4 WMMA accumulators).  Double-buffered LDS staging via
// GLOBAL_LOAD_ASYNC_TO_LDS_B128 (ASYNCcnt): stage k+1's tiles stream into the
// alternate buffer while stage k's WMMAs run.  80-byte row pitch keeps b128
// ds reads 16B-aligned and bank-conflict-free.
// MODE 0: bf16 out row-major.  MODE 1: bf16 out scattered to V^T[b,h,d,m].
// MODE 2: fp32 out = sigmoid(gate_logit_bf16) * acc.
template <int MODE>
__global__ __launch_bounds__(256) void gemm_bf16_kernel(
    const u16* __restrict__ A, const u16* __restrict__ W,
    void* __restrict__ Cout, const u16* __restrict__ Gate,
    int M, int N, int K) {
  constexpr int LDT = 40;                       // u16 pitch (80 B)
  constexpr uint32_t BUFB = 128 * LDT * 2;      // bytes per buffer (10240)
  __shared__ __align__(16) u16 As[2][128 * LDT];
  __shared__ __align__(16) u16 Bs[2][128 * LDT];
  const int nTiles = N >> 7;
  const int mBase = (blockIdx.x / nTiles) << 7;
  const int nBase = (blockIdx.x % nTiles) << 7;
  const int tid  = threadIdx.x;
  const int lane = tid & 31;
  const int w    = tid >> 5;
  const int wm   = (w & 3) * 32;
  const int wn   = (w >> 2) * 64;
  const int lr   = tid >> 1;          // staging row 0..127
  const int lh   = (tid & 1) << 4;    // staging col 0 or 16

  const u16* aRow = A + (size_t)(mBase + lr) * K + lh;
  const u16* bRow = W + (size_t)(nBase + lr) * K + lh;
  const uint32_t ldsA = lds_addr_of(&As[0][lr * LDT + lh]);
  const uint32_t ldsB = lds_addr_of(&Bs[0][lr * LDT + lh]);

  v8f acc[2][4];
  #pragma unroll
  for (int i = 0; i < 2; ++i)
    #pragma unroll
    for (int j = 0; j < 4; ++j) acc[i][j] = vzero();

  // prologue: stage k-step 0 into buffer 0
  async_g2l_b128(ldsA,      aRow);
  async_g2l_b128(ldsA + 16, aRow + 8);
  async_g2l_b128(ldsB,      bRow);
  async_g2l_b128(ldsB + 16, bRow + 8);

  const int nk = K >> 5;
  for (int ki = 0; ki < nk; ++ki) {
    const int p = ki & 1;
    if (ki + 1 < nk) {   // prefetch next stage into the other buffer
      const uint32_t bo = (uint32_t)(1 - p) * BUFB;
      const int ko = (ki + 1) << 5;
      async_g2l_b128(ldsA + bo,      aRow + ko);
      async_g2l_b128(ldsA + bo + 16, aRow + ko + 8);
      async_g2l_b128(ldsB + bo,      bRow + ko);
      async_g2l_b128(ldsB + bo + 16, bRow + ko + 8);
      wait_async_le4();      // async ops retire in order -> stage ki complete
    } else {
      wait_async_le0();
    }
    __syncthreads();         // all waves' stage-ki data visible in LDS

    Frag af[2];
    af[0] = load_a_frag(&As[p][(size_t)wm * LDT], LDT, lane);
    af[1] = load_a_frag(&As[p][(size_t)(wm + 16) * LDT], LDT, lane);
    Frag bf[4];
    #pragma unroll
    for (int j = 0; j < 4; ++j)
      bf[j] = load_b_frag(&Bs[p][(size_t)(wn + j * 16) * LDT], LDT, lane);
    #pragma unroll
    for (int i = 0; i < 2; ++i)
      #pragma unroll
      for (int j = 0; j < 4; ++j)
        acc[i][j] = wmma_bf16(af[i], bf[j], acc[i][j]);

    __syncthreads();         // buffer p may be overwritten next iteration
  }

  const int rowHalf = (lane < 16) ? 0 : 8;
  const int colL = lane & 15;
  #pragma unroll
  for (int i = 0; i < 2; ++i) {
    #pragma unroll
    for (int j = 0; j < 4; ++j) {
      const int nCol = nBase + wn + j * 16 + colL;
      #pragma unroll
      for (int r = 0; r < 8; ++r) {
        const int mRow = mBase + wm + i * 16 + r + rowHalf;
        const float val = acc[i][j][r];
        if (MODE == 0) {
          ((u16*)Cout)[(size_t)mRow * N + nCol] = f2bf(val);
        } else if (MODE == 1) {  // scatter to V^T[b][h][d][m], rows are b*256+m
          const int bb = mRow >> 8, mm = mRow & 255;
          const int hh = nCol >> 7, dd = nCol & 127;
          ((u16*)Cout)[((size_t)(bb * NHc + hh) * HDc + dd) * Mc + mm] = f2bf(val);
        } else {                 // gated fp32 output
          const float g  = bf2f(Gate[(size_t)mRow * N + nCol]);
          const float sg = 1.f / (1.f + __expf(-g));
          ((float*)Cout)[(size_t)mRow * N + nCol] = sg * val;
        }
      }
    }
  }
}

// ---------------------------------------------------------------- attention
// One block = (b, h, 32 q-rows), 4 waves. Phase 1: S = Q K^T (K=128) -> LDS.
// Phase 2: masked softmax over M=256 (4 lanes/row, shfl reductions).
// Phase 3: O = P V via WMMA (K=256), P from LDS, V^T from global (L2-resident).
__global__ __launch_bounds__(128) void mem_attn_kernel(
    const u16* __restrict__ Q, const u16* __restrict__ Km,
    const u16* __restrict__ VT, const u8* __restrict__ mask,
    u16* __restrict__ Out) {
  __shared__ float scores[32 * 256];
  __shared__ __align__(16) u16 probs[32 * 256];
  const int tid  = threadIdx.x;
  const int lane = tid & 31;
  const int w    = tid >> 5;
  const int qt = blockIdx.x & 127;
  const int h  = (blockIdx.x >> 7) & 15;
  const int b  = blockIdx.x >> 11;
  const int qRow0 = b * Lc + qt * 32;
  const int rowHalf = (lane < 16) ? 0 : 8;
  const int colL = lane & 15;
  const float NEG = -__builtin_inff();

  // ---- phase 1: scores
  #pragma unroll
  for (int t = 0; t < 8; ++t) {
    const int tile = w * 8 + t;
    const int mt = tile >> 4;    // q tile (0..1)
    const int nt = tile & 15;    // memory tile (0..15)
    v8f c = vzero();
    const u16* qb = Q  + (size_t)(qRow0 + mt * 16) * Hc + h * HDc;
    const u16* kb = Km + (size_t)(b * Mc + nt * 16) * Hc + h * HDc;
    #pragma unroll
    for (int kk = 0; kk < HDc; kk += 32) {
      Frag a  = load_a_frag(qb + kk, Hc, lane);
      Frag bb = load_b_frag(kb + kk, Hc, lane);  // k rows: d contiguous
      c = wmma_bf16(a, bb, c);
    }
    #pragma unroll
    for (int r = 0; r < 8; ++r) {
      const int lm = mt * 16 + r + rowHalf;
      const int gm = nt * 16 + colL;
      float s = c[r] * ATT_SCALE;
      if (!mask[b * Mc + gm]) s = NEG;
      scores[lm * 256 + gm] = s;
    }
  }
  __syncthreads();

  // ---- phase 2: softmax (4 lanes per row; rows align to lane groups of 4)
  {
    const int row = tid >> 2;
    const int i0  = (tid & 3) * 64;
    float* sr = &scores[row * 256 + i0];
    float mx = NEG;
    for (int i = 0; i < 64; ++i) mx = fmaxf(mx, sr[i]);
    mx = fmaxf(mx, __shfl_xor(mx, 1));
    mx = fmaxf(mx, __shfl_xor(mx, 2));
    float sum = 0.f;
    for (int i = 0; i < 64; ++i) {
      const float s = sr[i];
      const float e = (s == NEG) ? 0.f : __expf(s - mx);
      sr[i] = e;
      sum += e;
    }
    sum += __shfl_xor(sum, 1);
    sum += __shfl_xor(sum, 2);
    const float inv = (sum > 0.f) ? (1.f / sum) : 0.f;  // fully-masked -> zeros
    u16* pr = &probs[row * 256 + i0];
    for (int i = 0; i < 64; ++i) pr[i] = f2bf(sr[i] * inv);
  }
  __syncthreads();

  // ---- phase 3: O = P @ V   (per wave: 1 q-tile x 4 d-tiles)
  const int mt  = w >> 1;
  const int ntB = (w & 1) * 4;
  v8f acc[4];
  #pragma unroll
  for (int j = 0; j < 4; ++j) acc[j] = vzero();
  for (int kk = 0; kk < Mc; kk += 32) {
    Frag a = load_a_frag(&probs[mt * 16 * 256 + kk], 256, lane);
    #pragma unroll
    for (int j = 0; j < 4; ++j) {
      const u16* vb = VT + ((size_t)(b * NHc + h) * HDc + (ntB + j) * 16) * Mc + kk;
      Frag bb = load_b_frag(vb, Mc, lane);   // V^T rows: m contiguous
      acc[j] = wmma_bf16(a, bb, acc[j]);
    }
  }
  #pragma unroll
  for (int j = 0; j < 4; ++j) {
    const int col = h * HDc + (ntB + j) * 16 + colL;
    #pragma unroll
    for (int r = 0; r < 8; ++r) {
      const int gRow = qRow0 + mt * 16 + r + rowHalf;
      Out[(size_t)gRow * Hc + col] = f2bf(acc[j][r]);
    }
  }
}

// ---------------------------------------------------------------- launch
extern "C" void kernel_launch(void* const* d_in, const int* in_sizes, int n_in,
                              void* d_out, int out_size, void* d_ws, size_t ws_size,
                              hipStream_t stream) {
  const float* hs   = (const float*)d_in[0];     // [4,4096,2048]
  const float* mem  = (const float*)d_in[1];     // [4,256,2048]
  const u8*    mask = (const u8*)d_in[2];        // jax bool -> 1 byte
  const float* nw   = (const float*)d_in[3];     // [2048]
  const float* wq   = (const float*)d_in[4];
  const float* wk   = (const float*)d_in[5];
  const float* wv   = (const float*)d_in[6];
  const float* wo   = (const float*)d_in[7];
  const float* wg   = (const float*)d_in[8];

  const int NROW = Bc * Lc;   // 16384
  const int MROW = Bc * Mc;   // 1024

  char* ws = (char*)d_ws;
  size_t off = 0;
  auto alloc = [&](size_t bytes) {
    char* p = ws + off;
    off = (off + bytes + 255) & ~(size_t)255;
    return p;
  };
  u16* xb   = (u16*)alloc((size_t)NROW * Hc * 2);  // x bf16; later reused as attn-out
  u16* qb   = (u16*)alloc((size_t)NROW * Hc * 2);
  u16* gb   = (u16*)alloc((size_t)NROW * Hc * 2);  // gate logits bf16
  u16* wqb  = (u16*)alloc((size_t)Hc * Hc * 2);
  u16* wkb  = (u16*)alloc((size_t)Hc * Hc * 2);
  u16* wvb  = (u16*)alloc((size_t)Hc * Hc * 2);
  u16* wob  = (u16*)alloc((size_t)Hc * Hc * 2);
  u16* wgb  = (u16*)alloc((size_t)Hc * Hc * 2);
  u16* memb = (u16*)alloc((size_t)MROW * Hc * 2);
  u16* kbuf = (u16*)alloc((size_t)MROW * Hc * 2);
  u16* vTb  = (u16*)alloc((size_t)MROW * Hc * 2);

  // 1) bf16 conversions
  cvt_bf16_kernel<<<2048, 256, 0, stream>>>(wq, wqb, Hc * Hc);
  cvt_bf16_kernel<<<2048, 256, 0, stream>>>(wk, wkb, Hc * Hc);
  cvt_bf16_kernel<<<2048, 256, 0, stream>>>(wv, wvb, Hc * Hc);
  cvt_bf16_kernel<<<2048, 256, 0, stream>>>(wo, wob, Hc * Hc);
  cvt_bf16_kernel<<<2048, 256, 0, stream>>>(wg, wgb, Hc * Hc);
  cvt_bf16_kernel<<<2048, 256, 0, stream>>>(mem, memb, MROW * Hc);

  // 2) rmsnorm -> x bf16
  rmsnorm_kernel<<<NROW, 256, 0, stream>>>(hs, nw, xb);

  // 3) projections (gate + q first: they consume xb, which attention then reuses)
  const int gridBig   = (NROW / 128) * (Hc / 128);  // 2048
  const int gridSmall = (MROW / 128) * (Hc / 128);  // 128
  gemm_bf16_kernel<0><<<gridBig,   256, 0, stream>>>(xb,   wgb, gb,   nullptr, NROW, Hc, Hc);
  gemm_bf16_kernel<0><<<gridBig,   256, 0, stream>>>(xb,   wqb, qb,   nullptr, NROW, Hc, Hc);
  gemm_bf16_kernel<0><<<gridSmall, 256, 0, stream>>>(memb, wkb, kbuf, nullptr, MROW, Hc, Hc);
  gemm_bf16_kernel<1><<<gridSmall, 256, 0, stream>>>(memb, wvb, vTb,  nullptr, MROW, Hc, Hc);

  // 4) attention -> attn-out bf16 (reuses xb buffer)
  mem_attn_kernel<<<Bc * NHc * (Lc / 32), 128, 0, stream>>>(qb, kbuf, vTb, mask, xb);

  // 5) out = sigmoid(gate) * (attn_out @ wo^T), fp32
  gemm_bf16_kernel<2><<<gridBig, 256, 0, stream>>>(xb, wob, d_out, gb, NROW, Hc, Hc);
}
